// GPT_12386685682443
// MI455X (gfx1250) — compile-verified, matrix operand order checked by reading, not run on hardware
//
#include <hip/hip_runtime.h>
#include <hip/hip_bf16.h>

typedef _Float16 v8h  __attribute__((ext_vector_type(8)));
typedef _Float16 v16h __attribute__((ext_vector_type(16)));
typedef float    v8f  __attribute__((ext_vector_type(8)));
typedef int      v4i  __attribute__((vector_size(16)));

#define GB   2
#define GS   1024
#define GD   1024
#define GH   16
#define GHD  64
#define GL   6
#define GV   50257
#define GDFF 4096
#define GMT  (GB * GS)   // 2048 rows in the residual stream

enum { GF_BTRANS = 1, GF_BIAS = 2, GF_GELU = 4, GF_RES = 8, GF_OUTF16 = 16 };

// ---- CDNA5 async global->LDS copy (ASYNCcnt path), with sync fallback ------
#if defined(__has_builtin)
#if __has_builtin(__builtin_amdgcn_global_load_async_to_lds_b128)
#define HAVE_ASYNC_LDS 1
#endif
#if __has_builtin(__builtin_amdgcn_s_wait_asynccnt)
#define HAVE_WAIT_ASYNC_BUILTIN 1
#endif
#endif
#ifndef HAVE_ASYNC_LDS
#define HAVE_ASYNC_LDS 0
#endif
#ifndef HAVE_WAIT_ASYNC_BUILTIN
#define HAVE_WAIT_ASYNC_BUILTIN 0
#endif

__device__ __forceinline__ void async_copy_b128(const _Float16* g, _Float16* l)
{
#if HAVE_ASYNC_LDS
    __builtin_amdgcn_global_load_async_to_lds_b128(
        (v4i*)(void*)const_cast<_Float16*>(g),
        (__attribute__((address_space(3))) v4i*)l, 0, 0);
#else
    *(v8h*)l = *(const v8h*)g;
#endif
}

__device__ __forceinline__ void async_fence()
{
#if HAVE_ASYNC_LDS
#if HAVE_WAIT_ASYNC_BUILTIN
    __builtin_amdgcn_s_wait_asynccnt(0);
#else
    asm volatile("s_wait_asynccnt 0" ::: "memory");
#endif
#endif
}

// ---------------------------------------------------------------------------
// Generic batched GEMM (f16 operands, f32 accumulate):
//   C = alpha * A x B [+bias] [GELU] [+res]
// A: f16 [M,K] (lda).  B: f16 [K,N] (ldb) or, with GF_BTRANS, f16 [N,K] (ldb).
// C: f32 (optionally += res) or f16 with GF_OUTF16.
// Batch z = (zb, zh) = (z / Hn, z % Hn) with separate strides per operand.
//
// Block = 256 threads = 8 wave32.  Block tile 128x128, BK = 32.
// Waves 4(M) x 2(N); wave tile 32x64 -> 8 x v_wmma_f32_16x16x32_f16 / K-step.
// ---------------------------------------------------------------------------
__global__ __launch_bounds__(256)
void gemm_wmma_f16(const _Float16* __restrict__ A, const _Float16* __restrict__ Bm,
                   const float* __restrict__ bias, const float* __restrict__ res,
                   void* __restrict__ Cv,
                   int M, int N, int K, int lda, int ldb, int ldc,
                   long long sAb, long long sAh, long long sBb, long long sBh,
                   long long sCb, long long sCh, int Hn,
                   float alpha, int flags)
{
    __shared__ __align__(16) _Float16 As[128][40];  // 128 M-rows x 32 K (+8 pad)
    __shared__ __align__(16) _Float16 Bt[128][40];  // 128 N-rows x 32 K (B^T)

    const int tid  = threadIdx.x;
    const int lane = tid & 31;
    const int wave = tid >> 5;
    const int wm   = wave & 3;   // 4 waves along M -> 32 rows each
    const int wn   = wave >> 2;  // 2 waves along N -> 64 cols each

    const int bn0 = blockIdx.x * 128;
    const int bm0 = blockIdx.y * 128;
    const int z   = blockIdx.z;
    const int zb  = z / Hn;
    const int zh  = z % Hn;
    const long long offA = zb * sAb + zh * sAh;
    const long long offB = zb * sBb + zh * sBh;
    const long long offC = zb * sCb + zh * sCh;

    // CDNA5 16-bit fragment layout: lanes 0-15 hold K 0-7 & 16-23,
    // lanes 16-31 hold K 8-15 & 24-31 (same M/N row as lane&15).
    const int half = lane >> 4;
    const int lrow = lane & 15;
    const int kb   = half * 8;

    v8f acc[2][4];
    #pragma unroll
    for (int i = 0; i < 2; ++i)
        #pragma unroll
        for (int j = 0; j < 4; ++j) acc[i][j] = (v8f){};

    // staging coordinates: 128 rows x 32 K halves; 16 halves (32B) per thread
    const int srow = tid >> 1;          // 0..127
    const int sk   = (tid & 1) * 16;    // 0 or 16

    for (int k0 = 0; k0 < K; k0 += 32) {
        // ---- stage A tile (128x32 f16): two async b128 per thread
        {
            const int gm = bm0 + srow;
            if (gm < M) {
                const _Float16* src = A + offA + (long long)gm * lda + (k0 + sk);
                async_copy_b128(src,     &As[srow][sk]);
                async_copy_b128(src + 8, &As[srow][sk + 8]);
            } else {
                *(v8h*)&As[srow][sk]     = (v8h){};
                *(v8h*)&As[srow][sk + 8] = (v8h){};
            }
        }
        // ---- stage B tile transposed into Bt[N][K]
        if (flags & GF_BTRANS) {
            const int gn = bn0 + srow;
            if (gn < N) {
                const _Float16* src = Bm + offB + (long long)gn * ldb + (k0 + sk);
                async_copy_b128(src,     &Bt[srow][sk]);
                async_copy_b128(src + 8, &Bt[srow][sk + 8]);
            } else {
                *(v8h*)&Bt[srow][sk]     = (v8h){};
                *(v8h*)&Bt[srow][sk + 8] = (v8h){};
            }
        } else {
            const int kk = tid >> 3;          // 0..31
            const int nc = (tid & 7) * 16;    // 0..112
            const _Float16* src = Bm + offB + (long long)(k0 + kk) * ldb + (bn0 + nc);
            if (k0 + 32 < K) __builtin_prefetch(src + 32LL * ldb, 0, 1);
            #pragma unroll
            for (int i = 0; i < 16; ++i) {
                const int gn = bn0 + nc + i;
                Bt[nc + i][kk] = (gn < N) ? src[i] : (_Float16)0.0f;
            }
        }
        async_fence();
        __syncthreads();

        // ---- fragments from LDS
        v16h afrag[2], bfrag[4];
        #pragma unroll
        for (int i = 0; i < 2; ++i) {
            const int r = wm * 32 + i * 16 + lrow;
            v8h lo = *(const v8h*)&As[r][kb];
            v8h hi = *(const v8h*)&As[r][kb + 16];
            #pragma unroll
            for (int e = 0; e < 8; ++e) { afrag[i][e] = lo[e]; afrag[i][e + 8] = hi[e]; }
        }
        #pragma unroll
        for (int j = 0; j < 4; ++j) {
            const int r = wn * 64 + j * 16 + lrow;
            v8h lo = *(const v8h*)&Bt[r][kb];
            v8h hi = *(const v8h*)&Bt[r][kb + 16];
            #pragma unroll
            for (int e = 0; e < 8; ++e) { bfrag[j][e] = lo[e]; bfrag[j][e + 8] = hi[e]; }
        }

        #pragma unroll
        for (int i = 0; i < 2; ++i)
            #pragma unroll
            for (int j = 0; j < 4; ++j)
                acc[i][j] = __builtin_amdgcn_wmma_f32_16x16x32_f16(
                    false, afrag[i], false, bfrag[j], (short)0, acc[i][j], false, false);
        __syncthreads();
    }

    // ---- epilogue: C/D layout -> lane&15 = N column, half selects M 0-7 / 8-15
    #pragma unroll
    for (int i = 0; i < 2; ++i) {
        const int mb = bm0 + wm * 32 + i * 16 + half * 8;
        #pragma unroll
        for (int j = 0; j < 4; ++j) {
            const int gn = bn0 + wn * 64 + j * 16 + lrow;
            if (gn >= N) continue;
            #pragma unroll
            for (int r = 0; r < 8; ++r) {
                const int gm = mb + r;
                if (gm >= M) continue;
                float val = acc[i][j][r] * alpha;
                if (flags & GF_BIAS) val += bias[gn];
                if (flags & GF_GELU) val = 0.5f * val * (1.0f + erff(val * 0.70710678118654752f));
                const long long idx = offC + (long long)gm * ldc + gn;
                if (flags & GF_OUTF16) {
                    ((_Float16*)Cv)[idx] = (_Float16)val;
                } else {
                    if (flags & GF_RES) val += res[idx];
                    ((float*)Cv)[idx] = val;
                }
            }
        }
    }
}

// ---------------------------------------------------------------------------
// fp32 -> f16 conversion (n must be a multiple of 1024; grid = n/1024)
// ---------------------------------------------------------------------------
__global__ __launch_bounds__(256)
void cvt_f32_f16_kernel(const float* __restrict__ s, _Float16* __restrict__ d)
{
    const long long i = ((long long)blockIdx.x * 256 + threadIdx.x) * 4;
    const float4 v = *(const float4*)(s + i);
    d[i + 0] = (_Float16)v.x;
    d[i + 1] = (_Float16)v.y;
    d[i + 2] = (_Float16)v.z;
    d[i + 3] = (_Float16)v.w;
}

// ---------------------------------------------------------------------------
// x[b,s,:] = tok_emb[tokens[b,s],:] + pos_emb[s,:]   (fp32 residual stream)
// ---------------------------------------------------------------------------
__global__ __launch_bounds__(256)
void embed_kernel(const int* __restrict__ tokens, const float* __restrict__ tok_emb,
                  const float* __restrict__ pos_emb, float* __restrict__ x)
{
    const int row = blockIdx.x;          // 0 .. B*S-1
    const int s   = row % GS;
    const int tok = tokens[row];
    const float* te = tok_emb + (long long)tok * GD;
    const float* pe = pos_emb + (long long)s * GD;
    float* xr = x + (long long)row * GD;
    for (int d = threadIdx.x; d < GD; d += 256) xr[d] = te[d] + pe[d];
}

__device__ __forceinline__ float block_reduce_sum(float v, float* red)
{
    red[threadIdx.x] = v;
    __syncthreads();
    for (int st = 128; st > 0; st >>= 1) {
        if (threadIdx.x < st) red[threadIdx.x] += red[threadIdx.x + st];
        __syncthreads();
    }
    const float r = red[0];
    __syncthreads();
    return r;
}

__device__ __forceinline__ float block_reduce_max(float v, float* red)
{
    red[threadIdx.x] = v;
    __syncthreads();
    for (int st = 128; st > 0; st >>= 1) {
        if (threadIdx.x < st) red[threadIdx.x] = fmaxf(red[threadIdx.x], red[threadIdx.x + st]);
        __syncthreads();
    }
    const float r = red[0];
    __syncthreads();
    return r;
}

// ---------------------------------------------------------------------------
// Row LayerNorm over D=1024 (population variance, eps=1e-5); f16 output.
// ---------------------------------------------------------------------------
__global__ __launch_bounds__(256)
void layernorm_kernel(const float* __restrict__ x, const float* __restrict__ g,
                      const float* __restrict__ b, _Float16* __restrict__ y)
{
    __shared__ float red[256];
    const int row = blockIdx.x;
    const float* xr = x + (long long)row * GD;
    float s = 0.0f, s2 = 0.0f;
    for (int d = threadIdx.x; d < GD; d += 256) {
        const float v = xr[d];
        s += v; s2 += v * v;
    }
    const float mean = block_reduce_sum(s, red) * (1.0f / GD);
    const float var  = block_reduce_sum(s2, red) * (1.0f / GD) - mean * mean;
    const float inv  = rsqrtf(var + 1e-5f);
    _Float16* yr = y + (long long)row * GD;
    for (int d = threadIdx.x; d < GD; d += 256)
        yr[d] = (_Float16)((xr[d] - mean) * inv * g[d] + b[d]);
}

// ---------------------------------------------------------------------------
// Causal softmax: reads fp32 scores[B*H*S, S], writes f16 probs; j > i -> 0.
// ---------------------------------------------------------------------------
__global__ __launch_bounds__(256)
void softmax_causal_kernel(const float* __restrict__ scores, _Float16* __restrict__ probs)
{
    __shared__ float red[256];
    const long long row = blockIdx.x;     // 0 .. B*H*S-1
    const int i = (int)(row % GS);
    const float* p = scores + row * (long long)GS;
    _Float16* q = probs + row * (long long)GS;

    float mx = -3.402823466e38f;
    for (int j = threadIdx.x; j <= i; j += 256) mx = fmaxf(mx, p[j]);
    const float gmx = block_reduce_max(mx, red);

    float s = 0.0f;
    for (int j = threadIdx.x; j <= i; j += 256) s += __expf(p[j] - gmx);
    const float gs  = block_reduce_sum(s, red);
    const float inv = 1.0f / gs;

    for (int j = threadIdx.x; j < GS; j += 256)
        q[j] = (j <= i) ? (_Float16)(__expf(p[j] - gmx) * inv) : (_Float16)0.0f;
}

// ---------------------------------------------------------------------------
static inline void launch_gemm(hipStream_t st, const _Float16* A, const _Float16* Bm,
                               const float* bias, const float* res, void* Cm,
                               int M, int N, int K, int lda, int ldb, int ldc,
                               long long sAb, long long sAh, long long sBb, long long sBh,
                               long long sCb, long long sCh, int Hn, int zdim,
                               float alpha, int flags)
{
    dim3 grid((N + 127) / 128, (M + 127) / 128, zdim);
    gemm_wmma_f16<<<grid, dim3(256), 0, st>>>(A, Bm, bias, res, Cm,
                                              M, N, K, lda, ldb, ldc,
                                              sAb, sAh, sBb, sBh, sCb, sCh,
                                              Hn, alpha, flags);
}

static inline void launch_cvt(hipStream_t st, const float* s, _Float16* d, long long n)
{
    cvt_f32_f16_kernel<<<(unsigned)(n >> 10), 256, 0, st>>>(s, d);
}

extern "C" void kernel_launch(void* const* d_in, const int* in_sizes, int n_in,
                              void* d_out, int out_size, void* d_ws, size_t ws_size,
                              hipStream_t stream)
{
    const int*   tokens  = (const int*)  d_in[0];
    const float* tok_emb = (const float*)d_in[1];
    const float* pos_emb = (const float*)d_in[2];
    const float* ln1_g   = (const float*)d_in[3];
    const float* ln1_b   = (const float*)d_in[4];
    const float* wq      = (const float*)d_in[5];
    const float* wk      = (const float*)d_in[6];
    const float* wv      = (const float*)d_in[7];
    const float* wo      = (const float*)d_in[8];
    const float* ln2_g   = (const float*)d_in[9];
    const float* ln2_b   = (const float*)d_in[10];
    const float* w1      = (const float*)d_in[11];
    const float* b1      = (const float*)d_in[12];
    const float* w2      = (const float*)d_in[13];
    const float* b2      = (const float*)d_in[14];
    const float* lnf_g   = (const float*)d_in[15];
    const float* lnf_b   = (const float*)d_in[16];
    float* logits = (float*)d_out;

    // ---- workspace layout (~376 MB) ----
    const long long NX  = (long long)GMT * GD;          // 2,097,152
    const long long NSC = (long long)GB * GH * GS * GS; // 33,554,432
    const long long NY1 = (long long)GMT * GDFF;        // 8,388,608
    const long long NDD = (long long)GD * GD;           // 1,048,576
    const long long NW1 = (long long)GD * GDFF;         // 4,194,304
    const long long NEMB = (long long)GV * GD;          // 51,463,168

    float* x  = (float*)d_ws;          // fp32 residual stream
    float* sc = x + NX;                // fp32 attention scores
    _Float16* f16base = (_Float16*)(sc + NSC);
    _Float16* hh   = f16base;          // LN output        [GMT, D]
    _Float16* qbuf = hh   + NX;
    _Float16* kbuf = qbuf + NX;
    _Float16* vbuf = kbuf + NX;
    _Float16* obuf = vbuf + NX;
    _Float16* y1   = obuf + NX;        // MLP hidden       [GMT, 4D]
    _Float16* pp   = y1   + NY1;       // attn probs (f16) [B,H,S,S]
    _Float16* wbuf = pp   + NSC;       // per-layer f16 weights (rotating)
    _Float16* emb16 = wbuf + (4 * NDD + 2 * NW1);  // f16 tok_emb [V, D]

    _Float16* wq4 = wbuf;
    _Float16* wk4 = wq4 + NDD;
    _Float16* wv4 = wk4 + NDD;
    _Float16* wo4 = wv4 + NDD;
    _Float16* w14 = wo4 + NDD;
    _Float16* w24 = w14 + NW1;

    const float scale = 0.125f;                        // 1/sqrt(64)

    embed_kernel<<<GMT, 256, 0, stream>>>(tokens, tok_emb, pos_emb, x);
    launch_cvt(stream, tok_emb, emb16, NEMB);          // f16 tok_emb: 103 MB, L2-resident

    for (int l = 0; l < GL; ++l) {
        // convert this layer's weights to f16 (25 MB rotating buffer)
        launch_cvt(stream, wq + (long long)l * NDD, wq4, NDD);
        launch_cvt(stream, wk + (long long)l * NDD, wk4, NDD);
        launch_cvt(stream, wv + (long long)l * NDD, wv4, NDD);
        launch_cvt(stream, wo + (long long)l * NDD, wo4, NDD);
        launch_cvt(stream, w1 + (long long)l * NW1, w14, NW1);
        launch_cvt(stream, w2 + (long long)l * NW1, w24, NW1);
        const float* B1 = b1 + (long long)l * GDFF;
        const float* B2 = b2 + (long long)l * GD;

        // h = LN1(x)
        layernorm_kernel<<<GMT, 256, 0, stream>>>(x, ln1_g + l * GD, ln1_b + l * GD, hh);

        // q/k/v = h @ W{q,k,v}            (f16 in, f16 out)
        launch_gemm(stream, hh, wq4, nullptr, nullptr, qbuf, GMT, GD, GD, GD, GD, GD,
                    0, 0, 0, 0, 0, 0, 1, 1, 1.0f, GF_OUTF16);
        launch_gemm(stream, hh, wk4, nullptr, nullptr, kbuf, GMT, GD, GD, GD, GD, GD,
                    0, 0, 0, 0, 0, 0, 1, 1, 1.0f, GF_OUTF16);
        launch_gemm(stream, hh, wv4, nullptr, nullptr, vbuf, GMT, GD, GD, GD, GD, GD,
                    0, 0, 0, 0, 0, 0, 1, 1, 1.0f, GF_OUTF16);

        // scores[b,h] = scale * q[b,h] @ k[b,h]^T   (f32 out, batched over B*H)
        launch_gemm(stream, qbuf, kbuf, nullptr, nullptr, sc,
                    GS, GS, GHD, GD, GD, GS,
                    (long long)GS * GD, GHD,
                    (long long)GS * GD, GHD,
                    (long long)GH * GS * GS, (long long)GS * GS,
                    GH, GB * GH, scale, GF_BTRANS);

        softmax_causal_kernel<<<GB * GH * GS, 256, 0, stream>>>(sc, pp);

        // o[b,h] = P[b,h] @ v[b,h]        (f16 out)
        launch_gemm(stream, pp, vbuf, nullptr, nullptr, obuf,
                    GS, GHD, GS, GS, GD, GD,
                    (long long)GH * GS * GS, (long long)GS * GS,
                    (long long)GS * GD, GHD,
                    (long long)GS * GD, GHD,
                    GH, GB * GH, 1.0f, GF_OUTF16);

        // x = x + o @ Wo                  (fp32 residual update)
        launch_gemm(stream, obuf, wo4, nullptr, x, x, GMT, GD, GD, GD, GD, GD,
                    0, 0, 0, 0, 0, 0, 1, 1, 1.0f, GF_RES);

        // MLP: x = x + GELU(LN2(x) @ W1 + b1) @ W2 + b2
        layernorm_kernel<<<GMT, 256, 0, stream>>>(x, ln2_g + l * GD, ln2_b + l * GD, hh);
        launch_gemm(stream, hh, w14, B1, nullptr, y1, GMT, GDFF, GD, GD, GDFF, GDFF,
                    0, 0, 0, 0, 0, 0, 1, 1, 1.0f, GF_BIAS | GF_GELU | GF_OUTF16);
        launch_gemm(stream, y1, w24, B2, x, x, GMT, GD, GDFF, GDFF, GD, GD,
                    0, 0, 0, 0, 0, 0, 1, 1, 1.0f, GF_BIAS | GF_RES);
    }

    // final LN + tied logits head:  logits = LN_f(x) @ tok_emb^T  (fp32 out)
    layernorm_kernel<<<GMT, 256, 0, stream>>>(x, lnf_g, lnf_b, hh);
    launch_gemm(stream, hh, emb16, nullptr, nullptr, logits,
                GMT, GV, GD, GD, GD, GV,
                0, 0, 0, 0, 0, 0, 1, 1, 1.0f, GF_BTRANS);
}
